// TaeGAT_73366631350386
// MI455X (gfx1250) — compile-verified, matrix operand order checked by reading, not run on hardware
//
#include <hip/hip_runtime.h>
#include <hip/hip_bf16.h>

typedef __attribute__((ext_vector_type(16))) _Float16 v16h;
typedef __attribute__((ext_vector_type(8)))  _Float16 v8h;
typedef __attribute__((ext_vector_type(8)))  float    v8f;

union AFrag { v16h v; _Float16 e[16]; v8h h2[2]; };
union CFrag { v8f  v; float     e[8];  };

#define NEG_SLOPE 0.2f

// ---------- order-preserving float<->uint encoding for atomic max ----------
__device__ __forceinline__ unsigned fenc(float f) {
    unsigned u = __float_as_uint(f);
    return (u & 0x80000000u) ? ~u : (u | 0x80000000u);
}
__device__ __forceinline__ float fdec(unsigned k) {
    unsigned u = (k & 0x80000000u) ? (k ^ 0x80000000u) : ~k;
    return __uint_as_float(u);
}

// ---------- tiny prep: collapse edge attention to per-head scalars ----------
__global__ void prep_scalars(const float* We1, const float* ae1,
                             const float* We2, const float* ae2, float* ce) {
    int t = threadIdx.x;
    if (t < 2) {               // layer1, head t
        float s = 0.f;
        for (int c = 0; c < 64; ++c) s += We1[t * 64 + c] * ae1[t * 64 + c];
        ce[t] = s;
    } else if (t == 2) {       // layer2 single head
        float s = 0.f;
        for (int c = 0; c < 64; ++c) s += We2[c] * ae2[c];
        ce[2] = s;
    }
}

// ---------- transpose weight to column-major f16 for contiguous B fragments ----
// W is K x Ncols row-major (f32); Bt[col*K + k] = (f16) W[k*Ncols + col]
__global__ void transpose_w_f16(const float* __restrict__ W, _Float16* __restrict__ Bt,
                                int K, int Ncols) {
    int idx = blockIdx.x * blockDim.x + threadIdx.x;
    if (idx >= K * Ncols) return;
    int col = idx / K, k = idx % K;
    Bt[idx] = (_Float16)W[k * Ncols + col];
}

// ---------- degree + mean edge weight (self-loop attr) ----------
__global__ void degree_kernel(const int* dst, const float* w, float* cnt,
                              float* wsum, int E) {
    long e = (long)blockIdx.x * blockDim.x + threadIdx.x;
    if (e >= E) return;
    int d = dst[e];
    atomicAdd(&cnt[d], 1.0f);
    atomicAdd(&wsum[d], w[e]);
}
__global__ void loopattr_kernel(const float* cnt, float* wsum, int N) {
    int n = blockIdx.x * blockDim.x + threadIdx.x;
    if (n >= N) return;
    wsum[n] = wsum[n] / fmaxf(cnt[n], 1.0f);   // in-place -> loop_attr
}

// ---------- WMMA GEMM: C[N x NCOLS] = A[N x K] * B[K x NCOLS] ----------
// One wave32 per 16x16 output tile. A is f32 row-major (loaded as float4,
// packed to f16); B pre-transposed to col-major f16 (two b128 loads / step).
// Fragment layout per ISA 7.12.2 (16-bit A 16x32): lane = half*16 + m,
//   e[0..7]  <- K = ks + half*8 + {0..7}
//   e[8..15] <- K = ks + 16 + half*8 + {0..7}
// C/D layout: acc.e[v] = C[v + 8*half, lane&15].
template <int K, int NCOLS>
__global__ void wmma_gemm(const float* __restrict__ A, const int* __restrict__ gather,
                          const _Float16* __restrict__ Bt, float* __restrict__ C,
                          int N) {
    constexpr int NTN = NCOLS / 16;
    int lane = threadIdx.x & 31;
    int wave = threadIdx.x >> 5;
    int tile = blockIdx.x * (blockDim.x >> 5) + wave;
    int mt = tile / NTN, nt = tile % NTN;
    int mtiles = (N + 15) >> 4;
    if (mt >= mtiles) return;

    int half = lane >> 4;
    int l16  = lane & 15;
    int row  = mt * 16 + l16;
    int rowc = row < N ? row : N - 1;
    int arow = gather ? gather[rowc] : rowc;
    int col  = nt * 16 + l16;

    const float*    Arow = A + (size_t)arow * K;
    const _Float16* Bcol = Bt + (size_t)col * K;

    CFrag acc;
#pragma unroll
    for (int i = 0; i < 8; ++i) acc.e[i] = 0.f;

#pragma unroll
    for (int ks = 0; ks < K; ks += 32) {
        AFrag a, b;
        float4 a0 = *(const float4*)(Arow + ks + half * 8);
        float4 a1 = *(const float4*)(Arow + ks + half * 8 + 4);
        float4 a2 = *(const float4*)(Arow + ks + 16 + half * 8);
        float4 a3 = *(const float4*)(Arow + ks + 16 + half * 8 + 4);
        a.e[0]  = (_Float16)a0.x; a.e[1]  = (_Float16)a0.y;
        a.e[2]  = (_Float16)a0.z; a.e[3]  = (_Float16)a0.w;
        a.e[4]  = (_Float16)a1.x; a.e[5]  = (_Float16)a1.y;
        a.e[6]  = (_Float16)a1.z; a.e[7]  = (_Float16)a1.w;
        a.e[8]  = (_Float16)a2.x; a.e[9]  = (_Float16)a2.y;
        a.e[10] = (_Float16)a2.z; a.e[11] = (_Float16)a2.w;
        a.e[12] = (_Float16)a3.x; a.e[13] = (_Float16)a3.y;
        a.e[14] = (_Float16)a3.z; a.e[15] = (_Float16)a3.w;
        b.h2[0] = *(const v8h*)(Bcol + ks + half * 8);        // 16B-aligned
        b.h2[1] = *(const v8h*)(Bcol + ks + 16 + half * 8);
        acc.v = __builtin_amdgcn_wmma_f32_16x16x32_f16(
            /*neg_a=*/false, a.v, /*neg_b=*/false, b.v,
            /*c_mod=*/(short)0, acc.v, /*reuse_a=*/false, /*reuse_b=*/false);
    }

    float* Cbase = C + (size_t)(mt * 16 + 8 * half) * NCOLS + nt * 16 + l16;
    if (mt * 16 + 16 <= N) {                 // full tile: unguarded stores
#pragma unroll
        for (int v = 0; v < 8; ++v) Cbase[(size_t)v * NCOLS] = acc.e[v];
    } else {                                 // tail tile (not hit when N%16==0)
#pragma unroll
        for (int v = 0; v < 8; ++v) {
            if (mt * 16 + 8 * half + v < N) Cbase[(size_t)v * NCOLS] = acc.e[v];
        }
    }
}

// ---------- per-node attention coefficients ----------
__global__ void attn_coef(const float* __restrict__ h, const float* __restrict__ a_s,
                          const float* __restrict__ a_d, float* als, float* ald,
                          int N, int HD, int Hh) {
    int n = blockIdx.x * blockDim.x + threadIdx.x;
    if (n >= N) return;
    int Dd = HD / Hh;
    for (int hh = 0; hh < Hh; ++hh) {
        float s1 = 0.f, s2 = 0.f;
        for (int c = 0; c < Dd; ++c) {
            float v = h[(long)n * HD + hh * Dd + c];
            s1 += v * a_s[hh * Dd + c];
            s2 += v * a_d[hh * Dd + c];
        }
        als[(long)n * Hh + hh] = s1;
        ald[(long)n * Hh + hh] = s2;
    }
}

// ---------- pass A: leaky-relu logit, store, atomic max per (dst, head) ----------
__global__ void edge_logit_max(const int* src, const int* dst, const float* w,
                               const float* lattr, const float* als, const float* ald,
                               const float* ce, float* exbuf, unsigned* menc,
                               int E, int N, int Hh) {
    long e = (long)blockIdx.x * blockDim.x + threadIdx.x;
    long T = (long)E + N;
    if (e >= T) return;
    int s, d; float attr;
    if (e < E) { s = src[e]; d = dst[e]; attr = w[e]; }
    else       { s = d = (int)(e - E); attr = lattr[s]; }
    for (int hh = 0; hh < Hh; ++hh) {
        float l = als[(long)s * Hh + hh] + ald[(long)d * Hh + hh] + attr * ce[hh];
        l = l > 0.f ? l : NEG_SLOPE * l;
        exbuf[e * Hh + hh] = l;
        atomicMax(&menc[(long)d * Hh + hh], fenc(l));
    }
}

// ---------- pass B: exp(logit - max), store, atomic sum ----------
__global__ void edge_exp_sum(const int* src, const int* dst, float* exbuf,
                             const unsigned* menc, float* den, int E, int N, int Hh) {
    long e = (long)blockIdx.x * blockDim.x + threadIdx.x;
    long T = (long)E + N;
    if (e >= T) return;
    int d = (e < E) ? dst[e] : (int)(e - E);
    for (int hh = 0; hh < Hh; ++hh) {
        float l  = exbuf[e * Hh + hh];
        float mm = fdec(menc[(long)d * Hh + hh]);
        float ex = __expf(l - mm);
        exbuf[e * Hh + hh] = ex;
        atomicAdd(&den[(long)d * Hh + hh], ex);
    }
}

// ---------- pass C: one wave per edge, scatter alpha * h[s] into out[d] ----------
__global__ void edge_aggr(const int* src, const int* dst, const float* __restrict__ exbuf,
                          const float* __restrict__ den, const float* __restrict__ h,
                          float* out, int E, int N, int Hh, int HD) {
    int lane = threadIdx.x & 31;
    long e = (long)blockIdx.x * (blockDim.x >> 5) + (threadIdx.x >> 5);
    long T = (long)E + N;
    if (e >= T) return;
    int s, d;
    if (e < E) { s = src[e]; d = dst[e]; }
    else       { s = d = (int)(e - E); }
    int Dd = HD / Hh;
    int nper = HD >> 5;                       // 4 (layer1) or 2 (layer2)
    for (int i = 0; i < nper; ++i) {
        int c  = lane + (i << 5);
        int hh = c / Dd;
        float alpha = exbuf[e * Hh + hh] / (den[(long)d * Hh + hh] + 1e-16f);
        atomicAdd(&out[(long)d * HD + c], alpha * h[(long)s * HD + c]);
    }
}

// ---------- bias + LayerNorm (+ optional ReLU), one wave32 per node ----------
__global__ void bias_ln(const float* __restrict__ in, const float* __restrict__ bias,
                        const float* __restrict__ gamma, const float* __restrict__ beta,
                        float* out, int N, int C, int doRelu) {
    int lane = threadIdx.x & 31;
    long n = (long)blockIdx.x * (blockDim.x >> 5) + (threadIdx.x >> 5);
    if (n >= N) return;
    int nper = C >> 5;                        // <= 4
    float vals[4];
    float sum = 0.f;
    for (int i = 0; i < nper; ++i) {
        int c = lane + (i << 5);
        float v = in[n * C + c] + bias[c];
        vals[i] = v;
        sum += v;
    }
    for (int off = 16; off; off >>= 1) sum += __shfl_xor(sum, off);
    float mu = sum / (float)C;
    float vs = 0.f;
    for (int i = 0; i < nper; ++i) { float dd = vals[i] - mu; vs += dd * dd; }
    for (int off = 16; off; off >>= 1) vs += __shfl_xor(vs, off);
    float inv = rsqrtf(vs / (float)C + 1e-5f);
    for (int i = 0; i < nper; ++i) {
        int c = lane + (i << 5);
        float y = (vals[i] - mu) * inv * gamma[c] + beta[c];
        if (doRelu) y = fmaxf(y, 0.f);
        out[n * C + c] = y;
    }
}

extern "C" void kernel_launch(void* const* d_in, const int* in_sizes, int n_in,
                              void* d_out, int out_size, void* d_ws, size_t ws_size,
                              hipStream_t stream) {
    const int N = in_sizes[0];
    const int E = in_sizes[1] / 2;
    const long T = (long)E + N;
    const int HD1 = 128, D2 = 64;

    const int*   x    = (const int*)  d_in[0];
    const int*   src  = (const int*)  d_in[1];
    const int*   dst  = src + E;
    const float* ew   = (const float*)d_in[2];
    const float* emb  = (const float*)d_in[3];
    const float* W1   = (const float*)d_in[4];
    const float* as1  = (const float*)d_in[5];
    const float* ad1  = (const float*)d_in[6];
    const float* We1  = (const float*)d_in[7];
    const float* ae1  = (const float*)d_in[8];
    const float* b1   = (const float*)d_in[9];
    const float* g1   = (const float*)d_in[10];
    const float* bb1  = (const float*)d_in[11];
    const float* W2   = (const float*)d_in[12];
    const float* as2  = (const float*)d_in[13];
    const float* ad2  = (const float*)d_in[14];
    const float* We2  = (const float*)d_in[15];
    const float* ae2  = (const float*)d_in[16];
    const float* b2   = (const float*)d_in[17];
    const float* g2   = (const float*)d_in[18];
    const float* bb2  = (const float*)d_in[19];
    float* outp = (float*)d_out;

    // ---- workspace layout (zero-init block first: 200*N floats) ----
    size_t fN = (size_t)N;
    float*    ws    = (float*)d_ws;
    float*    cnt   = ws;                          // N
    float*    lattr = cnt + fN;                    // N   (wsum -> loop_attr)
    float*    den1  = lattr + fN;                  // 2N
    unsigned* m1    = (unsigned*)(den1 + 2 * fN);  // 2N  (encoded max)
    float*    den2  = (float*)(m1 + 2 * fN);       // N
    unsigned* m2    = (unsigned*)(den2 + fN);      // N
    float*    out1  = (float*)(m2 + fN);           // 128N
    float*    out2  = out1 + 128 * fN;             // 64N
    // end of zero block: 200N floats
    float*    h1    = out2 + 64 * fN;              // 128N
    float*    als1  = h1 + 128 * fN;               // 2N
    float*    ald1  = als1 + 2 * fN;               // 2N
    float*    h2    = ald1 + 2 * fN;               // 64N
    float*    als2  = h2 + 64 * fN;                // N
    float*    ald2  = als2 + fN;                   // N
    float*    exbuf = ald2 + fN;                   // 2T
    float*    ce    = exbuf + 2 * (size_t)T;       // 4
    // 16B-aligned f16 transposed weights
    size_t btoff = (((size_t)(ce + 4 - ws) + 3) & ~(size_t)3);
    _Float16* Bt1 = (_Float16*)(ws + btoff);       // 64*128 halfs (16 KB)
    _Float16* Bt2 = Bt1 + 64 * 128;                // 128*64 halfs

    hipMemsetAsync(d_ws, 0, 200 * fN * sizeof(float), stream);
    prep_scalars<<<1, 64, 0, stream>>>(We1, ae1, We2, ae2, ce);
    transpose_w_f16<<<(64 * 128 + 255) / 256, 256, 0, stream>>>(W1, Bt1, 64, 128);
    transpose_w_f16<<<(128 * 64 + 255) / 256, 256, 0, stream>>>(W2, Bt2, 128, 64);

    // degree / loop attr (shared by both layers)
    degree_kernel<<<(E + 255) / 256, 256, 0, stream>>>(dst, ew, cnt, lattr, E);
    loopattr_kernel<<<(N + 255) / 256, 256, 0, stream>>>(cnt, lattr, N);

    // ---------------- layer 1 ----------------
    {
        int mtiles = (N + 15) / 16, tiles = mtiles * (HD1 / 16);
        wmma_gemm<64, 128><<<(tiles + 7) / 8, 256, 0, stream>>>(emb, x, Bt1, h1, N);
    }
    attn_coef<<<(N + 255) / 256, 256, 0, stream>>>(h1, as1, ad1, als1, ald1, N, HD1, 2);
    edge_logit_max<<<(int)((T + 255) / 256), 256, 0, stream>>>(src, dst, ew, lattr,
                                                               als1, ald1, ce, exbuf, m1, E, N, 2);
    edge_exp_sum<<<(int)((T + 255) / 256), 256, 0, stream>>>(src, dst, exbuf, m1, den1, E, N, 2);
    edge_aggr<<<(int)((T + 7) / 8), 256, 0, stream>>>(src, dst, exbuf, den1, h1, out1, E, N, 2, HD1);
    bias_ln<<<(N + 7) / 8, 256, 0, stream>>>(out1, b1, g1, bb1, out1, N, HD1, 1);

    // ---------------- layer 2 ----------------
    {
        int mtiles = (N + 15) / 16, tiles = mtiles * (D2 / 16);
        wmma_gemm<128, 64><<<(tiles + 7) / 8, 256, 0, stream>>>(out1, nullptr, Bt2, h2, N);
    }
    attn_coef<<<(N + 255) / 256, 256, 0, stream>>>(h2, as2, ad2, als2, ald2, N, D2, 1);
    edge_logit_max<<<(int)((T + 255) / 256), 256, 0, stream>>>(src, dst, ew, lattr,
                                                               als2, ald2, ce + 2, exbuf, m2, E, N, 1);
    edge_exp_sum<<<(int)((T + 255) / 256), 256, 0, stream>>>(src, dst, exbuf, m2, den2, E, N, 1);
    edge_aggr<<<(int)((T + 7) / 8), 256, 0, stream>>>(src, dst, exbuf, den2, h2, out2, E, N, 1, D2);
    bias_ln<<<(N + 7) / 8, 256, 0, stream>>>(out2, b2, g2, bb2, outp, N, D2, 0);
}
